// DecoupledMVRowSelfAttnProcessor2_0_78194174591354
// MI455X (gfx1250) — matrix operand order repaired and checked by reading, not compiled, not run on hardware
//
#include <hip/hip_runtime.h>
#include <hip/hip_bf16.h>

// ---------------- problem constants (from setup_inputs) ----------------
#define B_   6
#define S_   1024
#define C_   1024
#define H_   16
#define D_   64
#define NV_  6
#define IH_  32
#define IW_  32
#define GMV_ (IH_ * H_)    // 512 groups for multi-view attention
#define SMV_ (NV_ * IW_)   // 192 seq length for multi-view attention
#define M_   (B_ * S_)     // 6144 rows

typedef unsigned short ushort_t;
typedef __attribute__((ext_vector_type(16))) __bf16        v16bf;
typedef __attribute__((ext_vector_type(8)))  float         v8f;
typedef __attribute__((ext_vector_type(4)))  unsigned int  u32x4;

__device__ __forceinline__ ushort_t f2bf(float f) {
  union { float f; unsigned int u; } x; x.f = f;
  unsigned int u = x.u;
  u += 0x7fffu + ((u >> 16) & 1u);      // round-to-nearest-even
  return (ushort_t)(u >> 16);
}
__device__ __forceinline__ float bf2f(ushort_t h) {
  union { float f; unsigned int u; } x; x.u = ((unsigned int)h) << 16;
  return x.f;
}

union Frag16 { v16bf v; u32x4 q[2]; };

// A-matrix 16x32 bf16 fragment (ISA 7.12.2): row = lane&15,
// K chunks {base..base+7, base+16..base+23}, base = (lane>>4)*8.
__device__ __forceinline__ v16bf load_a_frag(const ushort_t* mat, int ld,
                                             int r0, int k0, int lane) {
  const ushort_t* rp = mat + (size_t)(r0 + (lane & 15)) * ld
                           + k0 + ((lane >> 4) << 3);
  Frag16 f;
  f.q[0] = *(const u32x4*)(rp);
  f.q[1] = *(const u32x4*)(rp + 16);
  return f.v;
}

// B-matrix 32x16 bf16 fragment: col = lane&15, K chunks {base..base+7,
// base+16..base+23}, base = (lane>>4)*8. bt is the transposed operand laid out [n][k].
__device__ __forceinline__ v16bf load_b_frag(const ushort_t* bt, int ld,
                                             int n0, int k0, int lane) {
  const ushort_t* rp = bt + (size_t)(n0 + (lane & 15)) * ld
                           + k0 + ((lane >> 4) << 3);
  Frag16 f;
  f.q[0] = *(const u32x4*)(rp);
  f.q[1] = *(const u32x4*)(rp + 16);
  return f.v;
}

#define WMMA_BF16(a, b, c) \
  __builtin_amdgcn_wmma_f32_16x16x32_bf16(false, (a), false, (b), (short)0, (c), false, false)

// ---------------- elementwise / layout kernels ----------------
__global__ void cvt_bf16_k(const float* __restrict__ src, ushort_t* __restrict__ dst, int n) {
  int i = blockIdx.x * blockDim.x + threadIdx.x;
  if (i < n) dst[i] = f2bf(src[i]);
}

// W[K][N] f32 -> WT[N][K] bf16, K=N=1024, LDS-tiled for coalescing.
__global__ void wtrans_k(const float* __restrict__ W, ushort_t* __restrict__ WT) {
  __shared__ float tile[32][33];
  int n0 = blockIdx.x * 32, k0 = blockIdx.y * 32;
  for (int r = threadIdx.y; r < 32; r += 8)
    tile[r][threadIdx.x] = W[(size_t)(k0 + r) * C_ + n0 + threadIdx.x];
  __syncthreads();
  for (int r = threadIdx.y; r < 32; r += 8)
    WT[(size_t)(n0 + r) * C_ + k0 + threadIdx.x] = f2bf(tile[threadIdx.x][r]);
}

// out = residual + bo + bo_mv + bo_ref  (biases broadcast over rows)
__global__ void init_out_k(const float* __restrict__ hs, const float* __restrict__ bo,
                           const float* __restrict__ bo_mv, const float* __restrict__ bo_ref,
                           float* __restrict__ out, int n) {
  int i = blockIdx.x * blockDim.x + threadIdx.x;
  if (i >= n) return;
  int c = i & (C_ - 1);
  out[i] = hs[i] + bo[c] + bo_mv[c] + bo_ref[c];
}

// [b*S+s][h*64+d] -> [ (b*H+h) ][s][d], value scaled (scale=1 is exact identity)
__global__ void remap_heads_k(const ushort_t* __restrict__ src, ushort_t* __restrict__ dst,
                              float scale) {
  int i = blockIdx.x * blockDim.x + threadIdx.x;
  if (i >= M_ * C_) return;
  int row = i >> 10, col = i & 1023;
  int b = row >> 10, s = row & 1023;
  int h = col >> 6, d = col & 63;
  dst[((size_t)(b * H_ + h) * S_ + s) * D_ + d] = f2bf(bf2f(src[i]) * scale);
}

// multi-view regroup: b=view, s=i*32+w -> group g=i*H+h, pos p=view*32+w
__global__ void remap_mv_k(const ushort_t* __restrict__ src, ushort_t* __restrict__ dst,
                           float scale) {
  int i = blockIdx.x * blockDim.x + threadIdx.x;
  if (i >= M_ * C_) return;
  int row = i >> 10, col = i & 1023;
  int b = row >> 10, s = row & 1023;
  int h = col >> 6, d = col & 63;
  int ih = s >> 5, w = s & 31;
  int g = ih * H_ + h;
  int p = b * IW_ + w;
  dst[((size_t)g * SMV_ + p) * D_ + d] = f2bf(bf2f(src[i]) * scale);
}

// V transpose, base: [b*S+s][h*64+d] -> [g=b*H+h][d][s]; LDS-tiled 32x32.
// grid (S/32, D/32, G), block (32, 8)
__global__ void remap_heads_T_k(const ushort_t* __restrict__ src, ushort_t* __restrict__ dst) {
  __shared__ ushort_t t[32][33];
  int g = blockIdx.z;
  int b = g >> 4, h = g & 15;
  int s0 = blockIdx.x * 32, d0 = blockIdx.y * 32;
  int tx = threadIdx.x;
  for (int r = threadIdx.y; r < 32; r += 8)
    t[r][tx] = src[((size_t)b * S_ + s0 + r) * C_ + h * D_ + d0 + tx];
  __syncthreads();
  for (int r = threadIdx.y; r < 32; r += 8)
    dst[((size_t)g * D_ + d0 + r) * S_ + s0 + tx] = t[tx][r];
}

// V transpose, multi-view: src row = view*S + ih*32 + w, col = h*64+d
//  -> dst [g=ih*H+h][d][p=view*32+w].  grid (SMV/32, D/32, GMV), block (32, 8)
__global__ void remap_mv_T_k(const ushort_t* __restrict__ src, ushort_t* __restrict__ dst) {
  __shared__ ushort_t t[32][33];
  int g = blockIdx.z;
  int ih = g >> 4, h = g & 15;
  int p0 = blockIdx.x * 32, d0 = blockIdx.y * 32;
  int view = p0 >> 5;                 // p tile spans exactly one view
  int tx = threadIdx.x;
  for (int r = threadIdx.y; r < 32; r += 8)   // p = p0+r -> w = r
    t[r][tx] = src[((size_t)view * S_ + ih * 32 + r) * C_ + h * D_ + d0 + tx];
  __syncthreads();
  for (int r = threadIdx.y; r < 32; r += 8)
    dst[((size_t)g * D_ + d0 + r) * SMV_ + p0 + tx] = t[tx][r];
}

// attention out [g=b*H+h][s][d] -> row-major [b*S+s][h*64+d]
__global__ void remap_att_base_k(const ushort_t* __restrict__ src, ushort_t* __restrict__ dst) {
  int i = blockIdx.x * blockDim.x + threadIdx.x;
  if (i >= M_ * C_) return;
  int g = i >> 16;
  int s = (i >> 6) & 1023;
  int d = i & 63;
  int b = g >> 4, h = g & 15;
  dst[((size_t)b * S_ + s) * C_ + h * D_ + d] = src[i];
}

// mv attention out [g=ih*H+h][p=view*32+w][d] -> [view*S + ih*32+w][h*64+d]
__global__ void remap_att_mv_k(const ushort_t* __restrict__ src, ushort_t* __restrict__ dst) {
  int i = blockIdx.x * blockDim.x + threadIdx.x;
  if (i >= M_ * C_) return;
  int g = i / (SMV_ * D_);
  int rem = i - g * (SMV_ * D_);
  int p = rem >> 6, d = rem & 63;
  int ih = g >> 4, h = g & 15;
  int view = p >> 5, w = p & 31;
  int s = ih * 32 + w;
  dst[((size_t)view * S_ + s) * C_ + h * D_ + d] = src[i];
}

// ---------------- WMMA GEMM: C[M,N] (+)= A[M,K]bf16 x BT[N,K]bf16 ----------------
// 8 waves/block, each wave computes a 32x64 tile; k-loop software pipelined
// (next iteration's 6 fragments load while current 8 WMMAs execute).
template <int ACCUM>
__global__ __launch_bounds__(256)
void gemm_bf16_k(const ushort_t* __restrict__ A, const ushort_t* __restrict__ BT,
                 void* __restrict__ Cv, int M, int N, int K) {
  const int lane = threadIdx.x & 31;
  const int wid  = threadIdx.x >> 5;
  const int gw   = blockIdx.x * 8 + wid;
  const int ntile = N >> 6;
  const int mt = gw / ntile, nt = gw - mt * ntile;
  if (mt * 32 >= M) return;
  const int r0 = mt * 32, c0 = nt * 64;

  v8f acc[2][4];
  #pragma unroll
  for (int r = 0; r < 2; ++r)
    #pragma unroll
    for (int j = 0; j < 4; ++j)
      #pragma unroll
      for (int i = 0; i < 8; ++i) acc[r][j][i] = 0.f;

  v16bf a0 = load_a_frag(A, K, r0,      0, lane);
  v16bf a1 = load_a_frag(A, K, r0 + 16, 0, lane);
  v16bf b0 = load_b_frag(BT, K, c0,      0, lane);
  v16bf b1 = load_b_frag(BT, K, c0 + 16, 0, lane);
  v16bf b2 = load_b_frag(BT, K, c0 + 32, 0, lane);
  v16bf b3 = load_b_frag(BT, K, c0 + 48, 0, lane);

  for (int k0 = 32; k0 < K; k0 += 32) {
    v16bf na0 = load_a_frag(A, K, r0,      k0, lane);
    v16bf na1 = load_a_frag(A, K, r0 + 16, k0, lane);
    v16bf nb0 = load_b_frag(BT, K, c0,      k0, lane);
    v16bf nb1 = load_b_frag(BT, K, c0 + 16, k0, lane);
    v16bf nb2 = load_b_frag(BT, K, c0 + 32, k0, lane);
    v16bf nb3 = load_b_frag(BT, K, c0 + 48, k0, lane);

    acc[0][0] = WMMA_BF16(a0, b0, acc[0][0]);
    acc[1][0] = WMMA_BF16(a1, b0, acc[1][0]);
    acc[0][1] = WMMA_BF16(a0, b1, acc[0][1]);
    acc[1][1] = WMMA_BF16(a1, b1, acc[1][1]);
    acc[0][2] = WMMA_BF16(a0, b2, acc[0][2]);
    acc[1][2] = WMMA_BF16(a1, b2, acc[1][2]);
    acc[0][3] = WMMA_BF16(a0, b3, acc[0][3]);
    acc[1][3] = WMMA_BF16(a1, b3, acc[1][3]);

    a0 = na0; a1 = na1; b0 = nb0; b1 = nb1; b2 = nb2; b3 = nb3;
  }
  acc[0][0] = WMMA_BF16(a0, b0, acc[0][0]);
  acc[1][0] = WMMA_BF16(a1, b0, acc[1][0]);
  acc[0][1] = WMMA_BF16(a0, b1, acc[0][1]);
  acc[1][1] = WMMA_BF16(a1, b1, acc[1][1]);
  acc[0][2] = WMMA_BF16(a0, b2, acc[0][2]);
  acc[1][2] = WMMA_BF16(a1, b2, acc[1][2]);
  acc[0][3] = WMMA_BF16(a0, b3, acc[0][3]);
  acc[1][3] = WMMA_BF16(a1, b3, acc[1][3]);

  const int rh = (lane >> 4) << 3;    // C layout: row = i + 8*(lane>>4)
  const int cl = lane & 15;           //           col = lane & 15
  if (ACCUM) {
    float* C = (float*)Cv;
    #pragma unroll
    for (int r = 0; r < 2; ++r)
      #pragma unroll
      for (int j = 0; j < 4; ++j)
        #pragma unroll
        for (int i = 0; i < 8; ++i) {
          size_t idx = (size_t)(r0 + r * 16 + rh + i) * N + c0 + j * 16 + cl;
          C[idx] += acc[r][j][i];
        }
  } else {
    ushort_t* C = (ushort_t*)Cv;
    #pragma unroll
    for (int r = 0; r < 2; ++r)
      #pragma unroll
      for (int j = 0; j < 4; ++j)
        #pragma unroll
        for (int i = 0; i < 8; ++i) {
          size_t idx = (size_t)(r0 + r * 16 + rh + i) * N + c0 + j * 16 + cl;
          C[idx] = f2bf(acc[r][j][i]);
        }
  }
}

// ---------------- flash attention, online softmax through LDS ----------------
// Q (pre-scaled by 1/sqrt(d)*log2e), K: [G][Sq|Skv][64] bf16; Vt: [G][64][Skv];
// O: [G][Sq][64] bf16.  grid = (Sq/64, G), block = 128 (4 waves, 16 q-rows/wave)
__global__ __launch_bounds__(128)
void attn_k(const ushort_t* __restrict__ Q, const ushort_t* __restrict__ Kt,
            const ushort_t* __restrict__ Vt, ushort_t* __restrict__ O,
            int Sq, int Skv) {
  const int lane = threadIdx.x & 31;
  const int wid  = threadIdx.x >> 5;
  const int g    = blockIdx.y;
  const int q0   = blockIdx.x * 64 + wid * 16;

  const ushort_t* Qg = Q  + (size_t)g * Sq  * D_;
  const ushort_t* Kg = Kt + (size_t)g * Skv * D_;
  const ushort_t* Vg = Vt + (size_t)g * D_  * Skv;
  ushort_t*       Og = O  + (size_t)g * Sq  * D_;

  __shared__ float    sS[4][16][33];   // padded: lane halves hit distinct banks
  __shared__ ushort_t sP[4][16][40];   // stride 40 halves -> 80B rows, 16B aligned
  __shared__ float    sAl[4][16];
  __shared__ float    sL[4][16];

  const v16bf qf0 = load_a_frag(Qg, D_, q0, 0,  lane);
  const v16bf qf1 = load_a_frag(Qg, D_, q0, 32, lane);

  v8f o[4];
  #pragma unroll
  for (int j = 0; j < 4; ++j)
    #pragma unroll
    for (int i = 0; i < 8; ++i) o[j][i] = 0.f;

  float m_run = -__builtin_inff();
  float l_run = 0.f;
  const int rh = (lane >> 4) << 3;
  const int cl = lane & 15;

  for (int kv0 = 0; kv0 < Skv; kv0 += 32) {
    // ---- S = Q * K^T (16x32 tile): batch loads, then 4 wmma ----
    v16bf bk0 = load_b_frag(Kg, D_, kv0,      0,  lane);
    v16bf bk1 = load_b_frag(Kg, D_, kv0,      32, lane);
    v16bf bk2 = load_b_frag(Kg, D_, kv0 + 16, 0,  lane);
    v16bf bk3 = load_b_frag(Kg, D_, kv0 + 16, 32, lane);
    v8f s0, s1;
    #pragma unroll
    for (int i = 0; i < 8; ++i) { s0[i] = 0.f; s1[i] = 0.f; }
    s0 = WMMA_BF16(qf0, bk0, s0);
    s0 = WMMA_BF16(qf1, bk1, s0);
    s1 = WMMA_BF16(qf0, bk2, s1);
    s1 = WMMA_BF16(qf1, bk3, s1);

    // prefetch V fragments now: independent of softmax, overlap global
    // latency with LDS softmax + barriers
    v16bf bv0 = load_b_frag(Vg, Skv, 0,  kv0, lane);
    v16bf bv1 = load_b_frag(Vg, Skv, 16, kv0, lane);
    v16bf bv2 = load_b_frag(Vg, Skv, 32, kv0, lane);
    v16bf bv3 = load_b_frag(Vg, Skv, 48, kv0, lane);

    #pragma unroll
    for (int i = 0; i < 8; ++i) {
      sS[wid][rh + i][cl]      = s0[i];
      sS[wid][rh + i][16 + cl] = s1[i];
    }
    __syncthreads();

    // ---- online softmax (log2 domain; Q carried the scale) ----
    if (lane < 16) {
      float mx = m_run;
      #pragma unroll 4
      for (int j = 0; j < 32; ++j) mx = fmaxf(mx, sS[wid][lane][j]);
      float alpha = exp2f(m_run - mx);
      float lsum = 0.f;
      #pragma unroll 4
      for (int j = 0; j < 32; ++j) {
        float pv = exp2f(sS[wid][lane][j] - mx);
        lsum += pv;
        sP[wid][lane][j] = f2bf(pv);
      }
      l_run = l_run * alpha + lsum;
      m_run = mx;
      sAl[wid][lane] = alpha;
    }
    __syncthreads();

    // ---- rescale accumulators, O += P * V ----
    float al[8];
    #pragma unroll
    for (int i = 0; i < 8; ++i) al[i] = sAl[wid][rh + i];
    #pragma unroll
    for (int j = 0; j < 4; ++j)
      #pragma unroll
      for (int i = 0; i < 8; ++i) o[j][i] *= al[i];

    v16bf pf = load_a_frag(&sP[wid][0][0], 40, 0, 0, lane);
    o[0] = WMMA_BF16(pf, bv0, o[0]);
    o[1] = WMMA_BF16(pf, bv1, o[1]);
    o[2] = WMMA_BF16(pf, bv2, o[2]);
    o[3] = WMMA_BF16(pf, bv3, o[3]);
    __syncthreads();
  }

  if (lane < 16) sL[wid][lane] = l_run;
  __syncthreads();
  float li[8];
  #pragma unroll
  for (int i = 0; i < 8; ++i) li[i] = 1.f / sL[wid][rh + i];
  #pragma unroll
  for (int j = 0; j < 4; ++j)
    #pragma unroll
    for (int i = 0; i < 8; ++i)
      Og[(size_t)(q0 + rh + i) * D_ + j * 16 + cl] = f2bf(o[j][i] * li[i]);
}

// ---------------- host orchestration ----------------
extern "C" void kernel_launch(void* const* d_in, const int* in_sizes, int n_in,
                              void* d_out, int out_size, void* d_ws, size_t ws_size,
                              hipStream_t stream) {
  (void)in_sizes; (void)n_in; (void)out_size; (void)ws_size;
  const float* hs     = (const float*)d_in[0];
  const float* refh   = (const float*)d_in[1];
  const float* W[12]  = {
    (const float*)d_in[2],  (const float*)d_in[3],  (const float*)d_in[4],  (const float*)d_in[5],
    (const float*)d_in[7],  (const float*)d_in[8],  (const float*)d_in[9],  (const float*)d_in[10],
    (const float*)d_in[12], (const float*)d_in[13], (const float*)d_in[14], (const float*)d_in[15]
  };
  const float* bo     = (const float*)d_in[6];
  const float* bo_mv  = (const float*)d_in[11];
  const float* bo_ref = (const float*)d_in[16];
  float* out = (float*)d_out;

  const size_t NELT = (size_t)M_ * C_;        // 6,291,456 elements
  ushort_t* p    = (ushort_t*)d_ws;           // total ws use ~126 MB
  ushort_t* hsb  = p; p += NELT;
  ushort_t* refb = p; p += NELT;
  ushort_t* wT   = p; p += 12 * (size_t)C_ * C_;
  ushort_t* stg  = p; p += NELT;
  ushort_t* Qa   = p; p += NELT;
  ushort_t* Ka   = p; p += NELT;
  ushort_t* Va   = p; p += NELT;
  ushort_t* Ohb  = p; p += NELT;
  ushort_t* attR = p; p += NELT;

  const int EB = (int)((NELT + 255) / 256);
  const float sc = 0.125f * 1.4426950408889634f;   // 1/sqrt(64) * log2(e)
  const dim3 wtg(32, 32), wtb(32, 8);
  const dim3 vtg_base(S_ / 32, D_ / 32, B_ * H_);  // (32, 2, 96)
  const dim3 vtg_mv(SMV_ / 32, D_ / 32, GMV_);     // (6, 2, 512)
  const dim3 trb(32, 8);
  const int GEMM_BLOCKS = (M_ / 32) * (C_ / 64) / 8;  // 384

  cvt_bf16_k<<<EB, 256, 0, stream>>>(hs,   hsb,  (int)NELT);
  cvt_bf16_k<<<EB, 256, 0, stream>>>(refh, refb, (int)NELT);
  for (int w = 0; w < 12; ++w)
    wtrans_k<<<wtg, wtb, 0, stream>>>(W[w], wT + (size_t)w * C_ * C_);
  init_out_k<<<EB, 256, 0, stream>>>(hs, bo, bo_mv, bo_ref, out, (int)NELT);

  // ---- branch 0: base self-attention (weights 0..3) ----
  gemm_bf16_k<0><<<GEMM_BLOCKS, 256, 0, stream>>>(hsb, wT + 0 * (size_t)C_ * C_, stg, M_, C_, C_);
  remap_heads_k<<<EB, 256, 0, stream>>>(stg, Qa, sc);
  gemm_bf16_k<0><<<GEMM_BLOCKS, 256, 0, stream>>>(hsb, wT + 1 * (size_t)C_ * C_, stg, M_, C_, C_);
  remap_heads_k<<<EB, 256, 0, stream>>>(stg, Ka, 1.0f);
  gemm_bf16_k<0><<<GEMM_BLOCKS, 256, 0, stream>>>(hsb, wT + 2 * (size_t)C_ * C_, stg, M_, C_, C_);
  remap_heads_T_k<<<vtg_base, trb, 0, stream>>>(stg, Va);
  attn_k<<<dim3(S_ / 64, B_ * H_), 128, 0, stream>>>(Qa, Ka, Va, Ohb, S_, S_);
  remap_att_base_k<<<EB, 256, 0, stream>>>(Ohb, attR);
  gemm_bf16_k<1><<<GEMM_BLOCKS, 256, 0, stream>>>(attR, wT + 3 * (size_t)C_ * C_, out, M_, C_, C_);

  // ---- branch 1: multi-view row attention (weights 4..7) ----
  gemm_bf16_k<0><<<GEMM_BLOCKS, 256, 0, stream>>>(hsb, wT + 4 * (size_t)C_ * C_, stg, M_, C_, C_);
  remap_mv_k<<<EB, 256, 0, stream>>>(stg, Qa, sc);
  gemm_bf16_k<0><<<GEMM_BLOCKS, 256, 0, stream>>>(hsb, wT + 5 * (size_t)C_ * C_, stg, M_, C_, C_);
  remap_mv_k<<<EB, 256, 0, stream>>>(stg, Ka, 1.0f);
  gemm_bf16_k<0><<<GEMM_BLOCKS, 256, 0, stream>>>(hsb, wT + 6 * (size_t)C_ * C_, stg, M_, C_, C_);
  remap_mv_T_k<<<vtg_mv, trb, 0, stream>>>(stg, Va);
  attn_k<<<dim3(SMV_ / 64, GMV_), 128, 0, stream>>>(Qa, Ka, Va, Ohb, SMV_, SMV_);
  remap_att_mv_k<<<EB, 256, 0, stream>>>(Ohb, attR);
  gemm_bf16_k<1><<<GEMM_BLOCKS, 256, 0, stream>>>(attR, wT + 7 * (size_t)C_ * C_, out, M_, C_, C_);

  // ---- branch 2: reference attention (weights 8..11; K/V from ref states) ----
  gemm_bf16_k<0><<<GEMM_BLOCKS, 256, 0, stream>>>(hsb,  wT + 8  * (size_t)C_ * C_, stg, M_, C_, C_);
  remap_heads_k<<<EB, 256, 0, stream>>>(stg, Qa, sc);
  gemm_bf16_k<0><<<GEMM_BLOCKS, 256, 0, stream>>>(refb, wT + 9  * (size_t)C_ * C_, stg, M_, C_, C_);
  remap_heads_k<<<EB, 256, 0, stream>>>(stg, Ka, 1.0f);
  gemm_bf16_k<0><<<GEMM_BLOCKS, 256, 0, stream>>>(refb, wT + 10 * (size_t)C_ * C_, stg, M_, C_, C_);
  remap_heads_T_k<<<vtg_base, trb, 0, stream>>>(stg, Va);
  attn_k<<<dim3(S_ / 64, B_ * H_), 128, 0, stream>>>(Qa, Ka, Va, Ohb, S_, S_);
  remap_att_base_k<<<EB, 256, 0, stream>>>(Ohb, attR);
  gemm_bf16_k<1><<<GEMM_BLOCKS, 256, 0, stream>>>(attR, wT + 11 * (size_t)C_ * C_, out, M_, C_, C_);
}
